// Encoder_30502857736808
// MI455X (gfx1250) — matrix operand order repaired and checked by reading, not compiled, not run on hardware
//
#include <hip/hip_runtime.h>

typedef __bf16 bf16_t;
typedef __attribute__((ext_vector_type(16))) __bf16 v16bf;
typedef __attribute__((ext_vector_type(8)))  float  v8f;

#define B_  8
#define T_  4096
#define D_  1024
#define BT_ (B_ * T_)

// LDS row stride in bf16 elements for BK=64 tiles (144 bytes: 16B-aligned, skews banks)
#define LSTRIDE 72

union FragBF { v16bf v; uint4 q[2]; };

__device__ __forceinline__ int imin(int a, int b) { return a < b ? a : b; }

// ---------------------------------------------------------------------------
// f32 -> bf16 conversion (8 elements per thread, vectorized I/O)
// ---------------------------------------------------------------------------
__global__ __launch_bounds__(256) void cvt_bf16_kernel(const float* __restrict__ src,
                                                       bf16_t* __restrict__ dst,
                                                       long n) {
  long i = ((long)blockIdx.x * 256 + threadIdx.x) * 8;
  if (i + 8 > n) return;
  const float4* s4 = (const float4*)(src + i);
  float4 a = s4[0], b = s4[1];
  union { bf16_t h[8]; uint4 q; } o;
  o.h[0] = (bf16_t)a.x; o.h[1] = (bf16_t)a.y; o.h[2] = (bf16_t)a.z; o.h[3] = (bf16_t)a.w;
  o.h[4] = (bf16_t)b.x; o.h[5] = (bf16_t)b.y; o.h[6] = (bf16_t)b.z; o.h[7] = (bf16_t)b.w;
  *(uint4*)(dst + i) = o.q;
}

// ---------------------------------------------------------------------------
// Fused dual GEMM: xf = u @ WB^T, xb = u(shift+1) @ WBf^T
// Tile: BM=64 (65-row A tile for the +1 shift), BN=64, BK=64, 8 waves.
// Software-pipelined: round k+1's global tiles prefetched into registers
// while round k's 8 WMMAs run. Epilogue: boundary fixups, s = xf+xb ->
// d_out y region, row-dot partials for gamma logits via atomics.
// ---------------------------------------------------------------------------
__global__ __launch_bounds__(256) void gemm_kernel(
    const bf16_t* __restrict__ ub,     // [BT_][D_] bf16
    const bf16_t* __restrict__ wbb,    // [D_][D_]  bf16  (WB,  row = e/n, col = d/k)
    const bf16_t* __restrict__ wbfb,   // [D_][D_]  bf16  (WBf)
    const float*  __restrict__ u,      // [BT_][D_] f32 (boundary rows)
    const float*  __restrict__ w1,     // [D_]
    const float*  __restrict__ w2,     // [D_]
    float*        __restrict__ g1,     // [BT_] logit partials (pre-zeroed)
    float*        __restrict__ g2,     // [BT_]
    float*        __restrict__ yout)   // [BT_][D_] s = xf + xb
{
  __shared__ __align__(16) bf16_t lds[65 * LSTRIDE + 2 * 64 * LSTRIDE];
  bf16_t* Als  = lds;
  bf16_t* B0ls = lds + 65 * LSTRIDE;
  bf16_t* B1ls = B0ls + 64 * LSTRIDE;

  const int tid  = threadIdx.x;
  const int lane = tid & 31;
  const int wid  = tid >> 5;
  const int wm   = wid & 1;      // 2 M sub-blocks of 32 rows
  const int wn   = wid >> 1;     // 4 N sub-blocks of 16 cols
  const int half = lane >> 4;    // lane group (0: lanes 0-15, 1: lanes 16-31)
  const int lrow = lane & 15;

  const int nBase   = blockIdx.x * 64;   // output feature block
  const int rowBase = blockIdx.y * 64;   // (b,t) row block

  // ---- fixed per-thread staging assignment (no divergent fill loops) ----
  // A tile: 65 rows x 64 k = 520 uint4 chunks -> tid, tid+256, tail by tid<8
  const int arow = tid >> 3;             // 0..31
  const int acc8 = (tid & 7) * 8;        // k-offset within row (bf16 elems)
  const int gr0  = imin(rowBase + arow,      BT_ - 1);
  const int gr1  = imin(rowBase + arow + 32, BT_ - 1);
  const int grt  = imin(rowBase + 64,        BT_ - 1);   // shifted tail row (clamped)
  const bf16_t* apA0 = ub + (size_t)gr0 * D_ + acc8;
  const bf16_t* apA1 = ub + (size_t)gr1 * D_ + acc8;
  const bf16_t* apAt = ub + (size_t)grt * D_ + (tid & 7) * 8;  // only tid<8 uses
  // B tiles: 64 rows x 64 k = 512 chunks each -> tid, tid+256
  const bf16_t* bp00 = wbb  + (size_t)(nBase + arow)      * D_ + acc8;
  const bf16_t* bp01 = wbb  + (size_t)(nBase + arow + 32) * D_ + acc8;
  const bf16_t* bp10 = wbfb + (size_t)(nBase + arow)      * D_ + acc8;
  const bf16_t* bp11 = wbfb + (size_t)(nBase + arow + 32) * D_ + acc8;

  bf16_t* lsA0 = &Als[ arow       * LSTRIDE + acc8];
  bf16_t* lsA1 = &Als[(arow + 32) * LSTRIDE + acc8];
  bf16_t* lsAt = &Als[ 64         * LSTRIDE + (tid & 7) * 8];
  bf16_t* lsB00 = &B0ls[ arow       * LSTRIDE + acc8];
  bf16_t* lsB01 = &B0ls[(arow + 32) * LSTRIDE + acc8];
  bf16_t* lsB10 = &B1ls[ arow       * LSTRIDE + acc8];
  bf16_t* lsB11 = &B1ls[(arow + 32) * LSTRIDE + acc8];

  v8f accF0 = {}; v8f accF1 = {};
  v8f accB0 = {}; v8f accB1 = {};

  uint4 sA0, sA1, sAt, sB00, sB01, sB10, sB11;

  // prologue: stage round 0
  {
    const int k0 = 0;
    sA0  = *(const uint4*)(apA0 + k0);
    sA1  = *(const uint4*)(apA1 + k0);
    sB00 = *(const uint4*)(bp00 + k0);
    sB01 = *(const uint4*)(bp01 + k0);
    sB10 = *(const uint4*)(bp10 + k0);
    sB11 = *(const uint4*)(bp11 + k0);
    if (tid < 8) sAt = *(const uint4*)(apAt + k0);
  }

  const int aF0r = wm * 32 + lrow;        // fragment LDS row bases
  const int aF1r = aF0r + 16;
  const int bFr  = wn * 16 + lrow;

  for (int kk = 0; kk < D_ / 64; ++kk) {
    __syncthreads();                      // previous round's frags consumed
    *(uint4*)lsA0  = sA0;
    *(uint4*)lsA1  = sA1;
    *(uint4*)lsB00 = sB00;
    *(uint4*)lsB01 = sB01;
    *(uint4*)lsB10 = sB10;
    *(uint4*)lsB11 = sB11;
    if (tid < 8) *(uint4*)lsAt = sAt;
    __syncthreads();

    if (kk + 1 < D_ / 64) {               // prefetch next round during compute
      const int k0 = (kk + 1) * 64;
      sA0  = *(const uint4*)(apA0 + k0);
      sA1  = *(const uint4*)(apA1 + k0);
      sB00 = *(const uint4*)(bp00 + k0);
      sB01 = *(const uint4*)(bp01 + k0);
      sB10 = *(const uint4*)(bp10 + k0);
      sB11 = *(const uint4*)(bp11 + k0);
      if (tid < 8) sAt = *(const uint4*)(apAt + k0);
    }

#pragma unroll
    for (int ko = 0; ko < 64; ko += 32) {
      // A fragments: 16-bit A 16x32 layout (ISA 7.12.2):
      //   lanes 0-15: K 0..7 (v0-3), K 16..23 (v4-7); lanes 16-31: K 8..15, 24..31
      FragBF a0, a1, a0s, a1s, b0, b1;
      a0.q[0]  = *(const uint4*)&Als[ aF0r      * LSTRIDE + ko + half * 8];
      a0.q[1]  = *(const uint4*)&Als[ aF0r      * LSTRIDE + ko + 16 + half * 8];
      a1.q[0]  = *(const uint4*)&Als[ aF1r      * LSTRIDE + ko + half * 8];
      a1.q[1]  = *(const uint4*)&Als[ aF1r      * LSTRIDE + ko + 16 + half * 8];
      a0s.q[0] = *(const uint4*)&Als[(aF0r + 1) * LSTRIDE + ko + half * 8];
      a0s.q[1] = *(const uint4*)&Als[(aF0r + 1) * LSTRIDE + ko + 16 + half * 8];
      a1s.q[0] = *(const uint4*)&Als[(aF1r + 1) * LSTRIDE + ko + half * 8];
      a1s.q[1] = *(const uint4*)&Als[(aF1r + 1) * LSTRIDE + ko + 16 + half * 8];
      // B fragments: 16-bit B 32x16: lanes 0-15 hold K 0..7, 16..23; lanes 16-31 K 8..15, 24..31
      b0.q[0] = *(const uint4*)&B0ls[bFr * LSTRIDE + ko + half * 8];
      b0.q[1] = *(const uint4*)&B0ls[bFr * LSTRIDE + ko + 16 + half * 8];
      b1.q[0] = *(const uint4*)&B1ls[bFr * LSTRIDE + ko + half * 8];
      b1.q[1] = *(const uint4*)&B1ls[bFr * LSTRIDE + ko + 16 + half * 8];

      accF0 = __builtin_amdgcn_wmma_f32_16x16x32_bf16(false, a0.v,  false, b0.v, (short)0, accF0, false, false);
      accF1 = __builtin_amdgcn_wmma_f32_16x16x32_bf16(false, a1.v,  false, b0.v, (short)0, accF1, false, false);
      accB0 = __builtin_amdgcn_wmma_f32_16x16x32_bf16(false, a0s.v, false, b1.v, (short)0, accB0, false, false);
      accB1 = __builtin_amdgcn_wmma_f32_16x16x32_bf16(false, a1s.v, false, b1.v, (short)0, accB1, false, false);
    }
  }

  // Epilogue. C layout: VGPR v, lane l -> row = v + 8*(l>=16), col = l%16.
  const int gcol = nBase + wn * 16 + lrow;
#pragma unroll
  for (int t = 0; t < 2; ++t) {
    v8f aF = t ? accF1 : accF0;
    v8f aB = t ? accB1 : accB0;
    const int mbase = rowBase + wm * 32 + t * 16;
#pragma unroll
    for (int v = 0; v < 8; ++v) {
      const int grow = mbase + v + half * 8;
      const int tmod = grow & (T_ - 1);
      float xf = aF[v];
      float xb = aB[v];
      if (tmod == 0)      xf = u[(size_t)grow * D_ + gcol];  // xf[0]   = u[0]
      if (tmod == T_ - 1) xb = u[(size_t)grow * D_ + gcol];  // xb[T-1] = u[T-1]
      yout[(size_t)grow * D_ + gcol] = xf + xb;
      // partial row-dots for gamma logits; reduce over the 16 lanes of a row
      float p1 = xf * w1[gcol];
      float p2 = xb * w2[gcol];
      for (int m = 1; m < 16; m <<= 1) {
        p1 += __shfl_xor(p1, m, 32);
        p2 += __shfl_xor(p2, m, 32);
      }
      if (lrow == 0) {
        atomicAdd(&g1[grow], p1);
        atomicAdd(&g2[grow], p2);
      }
    }
  }
}

// ---------------------------------------------------------------------------
// Per-row epilogue: gamma, y (in place over s), final = sigmoid(y . Wc)
// One wave per row: 32 lanes x 32 elements.
// ---------------------------------------------------------------------------
__global__ __launch_bounds__(256) void rows_kernel(
    const float* __restrict__ u,
    const float* __restrict__ wc,
    const float* __restrict__ g1,
    const float* __restrict__ g2,
    const float* __restrict__ bptr,
    float* __restrict__ y,          // in: s = xf+xb ; out: y
    float* __restrict__ gamma_out,
    float* __restrict__ final_out)
{
  const int wid  = threadIdx.x >> 5;
  const int lane = threadIdx.x & 31;
  const int row  = blockIdx.x * 8 + wid;
  if (row >= BT_) return;

  const float glog = g1[row] + g2[row] + bptr[0];
  const float g    = 1.0f / (1.0f + __expf(-glog));
  const size_t base = (size_t)row * D_;

  float fl = 0.0f;
  for (int i = lane; i < D_; i += 32) {
    float s  = y[base + i];
    float uu = u[base + i];
    float yy = g * s + (1.0f - g) * uu;
    y[base + i] = yy;
    fl += yy * wc[i];
  }
  for (int m = 1; m < 32; m <<= 1) fl += __shfl_xor(fl, m, 32);
  if (lane == 0) {
    gamma_out[row] = g;
    final_out[row] = 1.0f / (1.0f + __expf(-fl));
  }
}

// ---------------------------------------------------------------------------
extern "C" void kernel_launch(void* const* d_in, const int* in_sizes, int n_in,
                              void* d_out, int out_size, void* d_ws, size_t ws_size,
                              hipStream_t stream) {
  (void)in_sizes; (void)n_in; (void)out_size; (void)ws_size;

  // setup_inputs order: u, WA, WB, WAf, WBf, w1, w2, b, Wc  (WA/WAf dead)
  const float* u   = (const float*)d_in[0];
  const float* WB  = (const float*)d_in[2];
  const float* WBf = (const float*)d_in[4];
  const float* w1  = (const float*)d_in[5];
  const float* w2  = (const float*)d_in[6];
  const float* b   = (const float*)d_in[7];
  const float* Wc  = (const float*)d_in[8];

  float* y_out     = (float*)d_out;              // (B,T,D)
  float* gamma_out = y_out + (size_t)BT_ * D_;   // (B,T,1)
  float* final_out = gamma_out + BT_;            // (B,T)

  // ws layout: g1[BT_] | g2[BT_] | u_bf16[BT_*D_] | WB_bf16[D_*D_] | WBf_bf16[D_*D_]
  char*   ws   = (char*)d_ws;
  float*  g1   = (float*)ws;
  float*  g2   = g1 + BT_;
  bf16_t* ub   = (bf16_t*)(ws + (size_t)2 * BT_ * sizeof(float));
  bf16_t* wbb  = ub  + (size_t)BT_ * D_;
  bf16_t* wbfb = wbb + (size_t)D_ * D_;

  hipMemsetAsync(g1, 0, (size_t)2 * BT_ * sizeof(float), stream);

  const long nu = (long)BT_ * D_;
  const long nw = (long)D_ * D_;
  cvt_bf16_kernel<<<(int)(nu / 8 / 256), 256, 0, stream>>>(u,   ub,   nu);
  cvt_bf16_kernel<<<(int)(nw / 8 / 256), 256, 0, stream>>>(WB,  wbb,  nw);
  cvt_bf16_kernel<<<(int)(nw / 8 / 256), 256, 0, stream>>>(WBf, wbfb, nw);

  dim3 grid(D_ / 64, BT_ / 64);   // 16 x 512
  gemm_kernel<<<grid, 256, 0, stream>>>(ub, wbb, wbfb, u, w1, w2, g1, g2, y_out);

  rows_kernel<<<BT_ / 8, 256, 0, stream>>>(u, Wc, g1, g2, b, y_out, gamma_out, final_out);
}
